// EmaVectorQuantizer_82703890252203
// MI455X (gfx1250) — compile-verified, matrix operand order checked by reading, not run on hardware
//
#include <hip/hip_runtime.h>
#include <hip/hip_bf16.h>

// ---- problem constants (from reference setup_inputs) ----
constexpr int VQ_N = 32768;
constexpr int VQ_K = 8192;
constexpr int VQ_D = 512;
constexpr float VQ_BETA = 0.25f;

typedef __attribute__((ext_vector_type(4)))  __bf16 v4bf;
typedef __attribute__((ext_vector_type(8)))  __bf16 v8bf;
typedef __attribute__((ext_vector_type(16))) __bf16 v16bf;
typedef __attribute__((ext_vector_type(8)))  float  v8f;
typedef int v4i __attribute__((vector_size(16)));

// CDNA5 async global->LDS copy path (ASYNCcnt-tracked), with fallback.
#if defined(__has_builtin)
#if __has_builtin(__builtin_amdgcn_global_load_async_to_lds_b128) && \
    __has_builtin(__builtin_amdgcn_s_wait_asynccnt)
#define VQ_ASYNC 1
#endif
#endif
#ifndef VQ_ASYNC
#define VQ_ASYNC 0
#endif

static __device__ __forceinline__ void vq_cp16_to_lds(const __bf16* g,
                                                      __bf16* l) {
#if VQ_ASYNC
  // Prototype (from hipcc diagnostic): (v4i AS1*, v4i AS3*, imm, imm)
  __builtin_amdgcn_global_load_async_to_lds_b128(
      (__attribute__((address_space(1))) v4i*)g,
      (__attribute__((address_space(3))) v4i*)l, 0, 0);
#else
  *reinterpret_cast<v8bf*>(l) = *reinterpret_cast<const v8bf*>(g);
#endif
}

static __device__ __forceinline__ void vq_wait_cp() {
#if VQ_ASYNC
  __builtin_amdgcn_s_wait_asynccnt(0);
#endif
}

static __device__ __forceinline__ v16bf cat8(v8bf lo, v8bf hi) {
  v16bf r;
#pragma unroll
  for (int i = 0; i < 8; ++i) { r[i] = lo[i]; r[8 + i] = hi[i]; }
  return r;
}

// ---------------------------------------------------------------------------
// Kernel 1: z (fp32) -> bf16, vectorized 4-wide.
// ---------------------------------------------------------------------------
__global__ __launch_bounds__(256) void vq_cvt_z_kernel(
    const float* __restrict__ z, __bf16* __restrict__ zb, int n4) {
  int i = blockIdx.x * 256 + threadIdx.x;
  if (i < n4) {
    float4 v = reinterpret_cast<const float4*>(z)[i];
    v4bf o;
    o[0] = (__bf16)v.x; o[1] = (__bf16)v.y;
    o[2] = (__bf16)v.z; o[3] = (__bf16)v.w;
    reinterpret_cast<v4bf*>(zb)[i] = o;
  }
}

// ---------------------------------------------------------------------------
// Kernel 2: codebook (fp32) -> bf16 + cnorm[k] = ||c_k||^2 (fp32).
// One block per codebook row.
// ---------------------------------------------------------------------------
__global__ __launch_bounds__(256) void vq_prep_cb_kernel(
    const float* __restrict__ cbf, __bf16* __restrict__ cbb,
    float* __restrict__ cnorm) {
  const int k = blockIdx.x;
  const int t = threadIdx.x;
  const float* row = cbf + (size_t)k * VQ_D;
  __bf16* orow = cbb + (size_t)k * VQ_D;
  float ss = 0.0f;
#pragma unroll
  for (int i = 0; i < VQ_D / 256; ++i) {
    int d = t + i * 256;
    float v = row[d];
    orow[d] = (__bf16)v;
    ss += v * v;
  }
#pragma unroll
  for (int m = 1; m < 32; m <<= 1) ss += __shfl_xor(ss, m, 32);
  __shared__ float wsum[8];
  if ((t & 31) == 0) wsum[t >> 5] = ss;
  __syncthreads();
  if (t == 0) {
    float tot = 0.0f;
#pragma unroll
    for (int w = 0; w < 8; ++w) tot += wsum[w];
    cnorm[k] = tot;
  }
}

// ---------------------------------------------------------------------------
// Kernel 3: fused bf16 WMMA GEMM (z @ cb^T) + running argmin over K.
// 256 threads = 8 waves; wave w owns rows [blk*128 + w*16, +16) with its A
// tile in registers. Per iteration a 32-column codebook tile is staged into
// double-buffered LDS (async global->LDS copies overlap compute); each wave
// runs two independent 16-WMMA accumulator chains (columns 0-15 / 16-31 of
// the tile) so ds_load latency hides behind the other chain's WMMA.
// Score = cnorm[col] - 2*dot (||z||^2 constant per row: dropped for argmin).
// ---------------------------------------------------------------------------
__global__ __launch_bounds__(256) void vq_argmin_kernel(
    const __bf16* __restrict__ zb, const __bf16* __restrict__ cb,
    const float* __restrict__ cnorm, int* __restrict__ idxout) {
  constexpr int PADH = VQ_D + 8;   // 520 halves/row -> conflict-free banks
  constexpr int BUFH = 32 * PADH;  // one 32-column tile
  constexpr int NT = VQ_K / 32;    // 256 tiles
  __shared__ __align__(16) __bf16 smemB[2 * BUFH];

  const int tid = threadIdx.x;
  const int lane = tid & 31;
  const int wave = tid >> 5;
  const int rowbase = blockIdx.x * 128 + wave * 16;
  const int klo = (lane & 16) ? 8 : 0;  // A/B hardware layout: hi half-wave
  const int mrow = rowbase + (lane & 15);

  // Prologue: stage tile 0 into buffer 0 (overlaps with A-fragment loads).
#pragma unroll
  for (int i = 0; i < 8; ++i) {
    int c = tid + i * 256;
    int row = c >> 6;  // 64 x v8bf chunks per row
    int col8 = c & 63;
    vq_cp16_to_lds(cb + (size_t)row * VQ_D + col8 * 8,
                   smemB + row * PADH + col8 * 8);
  }

  // Load A fragments (16 rows x 512) into registers in WMMA A layout.
  v16bf afrag[16];
#pragma unroll
  for (int dc = 0; dc < 16; ++dc) {
    const __bf16* p = zb + (size_t)mrow * VQ_D + dc * 32 + klo;
    v8bf lo = *reinterpret_cast<const v8bf*>(p);
    v8bf hi = *reinterpret_cast<const v8bf*>(p + 16);
    afrag[dc] = cat8(lo, hi);
  }

  float minv[16];
  int   mini[16];
#pragma unroll
  for (int j = 0; j < 16; ++j) { minv[j] = 3.4e38f; mini[j] = 0; }

  vq_wait_cp();
  __syncthreads();

  for (int t = 0; t < NT; ++t) {
    const int p = t & 1;
    const __bf16* bufp = smemB + p * BUFH;

    // Stage tile t+1 into the other buffer (async; overlaps WMMA below).
    if (t + 1 < NT) {
      const int cb1 = (t + 1) * 32;
      __bf16* dst = smemB + (1 - p) * BUFH;
#pragma unroll
      for (int i = 0; i < 8; ++i) {
        int c = tid + i * 256;
        int row = c >> 6;
        int col8 = c & 63;
        vq_cp16_to_lds(cb + (size_t)(cb1 + row) * VQ_D + col8 * 8,
                       dst + row * PADH + col8 * 8);
      }
    }

    // Two independent accumulator chains over the 32-column tile.
    v8f acc0 = {};
    v8f acc1 = {};
#pragma unroll
    for (int dc = 0; dc < 16; ++dc) {
      const __bf16* b0 = bufp + (lane & 15) * PADH + dc * 32 + klo;
      const __bf16* b1 = b0 + 16 * PADH;
      v16bf f0 = cat8(*reinterpret_cast<const v8bf*>(b0),
                      *reinterpret_cast<const v8bf*>(b0 + 16));
      v16bf f1 = cat8(*reinterpret_cast<const v8bf*>(b1),
                      *reinterpret_cast<const v8bf*>(b1 + 16));
      acc0 = __builtin_amdgcn_wmma_f32_16x16x32_bf16(
          false, afrag[dc], false, f0, (short)0, acc0, false, false);
      acc1 = __builtin_amdgcn_wmma_f32_16x16x32_bf16(
          false, afrag[dc], false, f1, (short)0, acc1, false, false);
    }

    const int col0 = t * 32 + (lane & 15);
    const int col1 = col0 + 16;
    const float cn0 = cnorm[col0];
    const float cn1 = cnorm[col1];
#pragma unroll
    for (int j = 0; j < 8; ++j) {
      float s0 = fmaf(-2.0f, acc0[j], cn0);
      bool l0 = s0 < minv[j];
      minv[j] = l0 ? s0 : minv[j];
      mini[j] = l0 ? col0 : mini[j];
      float s1 = fmaf(-2.0f, acc1[j], cn1);
      bool l1 = s1 < minv[8 + j];
      minv[8 + j] = l1 ? s1 : minv[8 + j];
      mini[8 + j] = l1 ? col1 : mini[8 + j];
    }

    vq_wait_cp();    // all of this thread's async copies into buf[1-p] done
    __syncthreads(); // everyone done reading buf[p] and staging buf[1-p]
  }

  // Merge the two column-subtile candidates (tie-break: smaller index),
  // then cross-lane argmin within each 16-lane half (C layout: lanes 0-15
  // hold row j, lanes 16-31 hold row j+8).
#pragma unroll
  for (int j = 0; j < 8; ++j) {
    float v1 = minv[8 + j];
    int i1 = mini[8 + j];
    if (v1 < minv[j] || (v1 == minv[j] && i1 < mini[j])) {
      minv[j] = v1; mini[j] = i1;
    }
#pragma unroll
    for (int m = 1; m <= 8; m <<= 1) {
      float ov = __shfl_xor(minv[j], m, 32);
      int   oi = __shfl_xor(mini[j], m, 32);
      if (ov < minv[j] || (ov == minv[j] && oi < mini[j])) {
        minv[j] = ov; mini[j] = oi;
      }
    }
  }
  if (lane == 0) {
#pragma unroll
    for (int j = 0; j < 8; ++j) idxout[rowbase + j] = mini[j];
  } else if (lane == 16) {
#pragma unroll
    for (int j = 0; j < 8; ++j) idxout[rowbase + 8 + j] = mini[j];
  }
}

// ---------------------------------------------------------------------------
// Kernel 4: gather z_q, write z_st = z + (z_q - z), index-as-float, and
// deterministic per-row squared-error partial sums.
// ---------------------------------------------------------------------------
__global__ __launch_bounds__(256) void vq_gather_kernel(
    const float* __restrict__ z, const float* __restrict__ codebook,
    const int* __restrict__ idx, float* __restrict__ out,
    float* __restrict__ partials) {
  const int n = blockIdx.x;
  const int t = threadIdx.x;
  const int k = idx[n];
  const float* crow = codebook + (size_t)k * VQ_D;
  const float* zrow = z + (size_t)n * VQ_D;
  float* orow = out + (size_t)n * VQ_D;
  float ss = 0.0f;
#pragma unroll
  for (int i = 0; i < VQ_D / 256; ++i) {
    int d = t + i * 256;
    float zq = crow[d];
    float zv = zrow[d];
    orow[d] = zv + (zq - zv);  // straight-through: same fp ops as reference
    float e = zv - zq;
    ss += e * e;
  }
#pragma unroll
  for (int m = 1; m < 32; m <<= 1) ss += __shfl_xor(ss, m, 32);
  __shared__ float wsum[8];
  if ((t & 31) == 0) wsum[t >> 5] = ss;
  __syncthreads();
  if (t == 0) {
    float tot = 0.0f;
#pragma unroll
    for (int w = 0; w < 8; ++w) tot += wsum[w];
    partials[n] = tot;
    out[(size_t)VQ_N * VQ_D + n] = (float)k;  // indices output slot
  }
}

// ---------------------------------------------------------------------------
// Kernel 5: deterministic final loss reduction (single block).
// ---------------------------------------------------------------------------
__global__ __launch_bounds__(256) void vq_loss_kernel(
    const float* __restrict__ partials, float* __restrict__ out_loss) {
  float s = 0.0f;
  for (int i = threadIdx.x; i < VQ_N; i += 256) s += partials[i];
#pragma unroll
  for (int m = 1; m < 32; m <<= 1) s += __shfl_xor(s, m, 32);
  __shared__ float wsum[8];
  if ((threadIdx.x & 31) == 0) wsum[threadIdx.x >> 5] = s;
  __syncthreads();
  if (threadIdx.x == 0) {
    float tot = 0.0f;
#pragma unroll
    for (int w = 0; w < 8; ++w) tot += wsum[w];
    *out_loss = VQ_BETA * tot / ((float)VQ_N * (float)VQ_D);
  }
}

// ---------------------------------------------------------------------------
extern "C" void kernel_launch(void* const* d_in, const int* in_sizes, int n_in,
                              void* d_out, int out_size, void* d_ws,
                              size_t ws_size, hipStream_t stream) {
  const float* z = (const float*)d_in[0];         // [N, D]
  const float* codebook = (const float*)d_in[1];  // [K, D]
  float* out = (float*)d_out;  // [N*D] z_st | [N] idx | [1] loss

  // workspace layout (bytes)
  char* ws = (char*)d_ws;
  __bf16* zb     = (__bf16*)(ws);                                   // 32 MB
  __bf16* cbb    = (__bf16*)(ws + (size_t)VQ_N * VQ_D * 2);         //  8 MB
  float*  cnorm  = (float*)(ws + (size_t)(VQ_N + VQ_K) * VQ_D * 2); // 32 KB
  int*    idxbuf = (int*)((char*)cnorm + (size_t)VQ_K * 4);         // 128 KB
  float*  parts  = (float*)((char*)idxbuf + (size_t)VQ_N * 4);      // 128 KB

  // 1) z -> bf16
  {
    int n4 = VQ_N * VQ_D / 4;
    vq_cvt_z_kernel<<<(n4 + 255) / 256, 256, 0, stream>>>(z, zb, n4);
  }
  // 2) codebook -> bf16 + cnorm
  vq_prep_cb_kernel<<<VQ_K, 256, 0, stream>>>(codebook, cbb, cnorm);
  // 3) fused WMMA GEMM + argmin
  vq_argmin_kernel<<<VQ_N / 128, 256, 0, stream>>>(zb, cbb, cnorm, idxbuf);
  // 4) gather + z_st + partial losses
  vq_gather_kernel<<<VQ_N, 256, 0, stream>>>(z, codebook, idxbuf, out, parts);
  // 5) final loss
  vq_loss_kernel<<<1, 256, 0, stream>>>(parts, out + (size_t)VQ_N * VQ_D + VQ_N);
}
